// EPGRUCell_28948079575389
// MI455X (gfx1250) — matrix operand order repaired and checked by reading, compiled, not run on hardware
//
#include <hip/hip_runtime.h>
#include <stdint.h>

#define B_ 64
#define T_ 512
#define D_ 256
#define H_ 512

typedef __attribute__((ext_vector_type(16))) __bf16        v16bf;
typedef __attribute__((ext_vector_type(8)))  float         v8f;
typedef __attribute__((ext_vector_type(8)))  unsigned int  v8u;

// ---------- scalar helpers ----------
__device__ __forceinline__ unsigned short f2bf(float f) {
  unsigned int u = __builtin_bit_cast(unsigned int, f);
  u += 0x7fffu + ((u >> 16) & 1u);               // RNE truncate to bf16
  return (unsigned short)(u >> 16);
}
__device__ __forceinline__ unsigned int pk2bf(float lo, float hi) {
  return (unsigned int)f2bf(lo) | ((unsigned int)f2bf(hi) << 16);
}
__device__ __forceinline__ float sigmoidf_(float x) {
  return 1.0f / (1.0f + __expf(-x));
}

// ---------- WMMA fragment loaders (wave32, 16x16x32 bf16) ----------
// A (16x32, row-major bf16): lane l -> row m=l&15, half=l>>4.
// VGPRs 0-3 hold K=[kc+half*8 .. +7], VGPRs 4-7 hold K=[kc+16+half*8 .. +7].
__device__ __forceinline__ v16bf load_a_frag(const unsigned short* A, int lda, int row0, int kc) {
  int lane = threadIdx.x & 31;
  const unsigned short* p = A + (size_t)(row0 + (lane & 15)) * lda + kc + ((lane >> 4) << 3);
  uint4 lo = *reinterpret_cast<const uint4*>(p);
  uint4 hi = *reinterpret_cast<const uint4*>(p + 16);
  v8u u = { lo.x, lo.y, lo.z, lo.w, hi.x, hi.y, hi.z, hi.w };
  return __builtin_bit_cast(v16bf, u);
}

// B (32x16) pre-packed fragment-major: [(kt*NT+nt)*32 + lane] * 8 uint32s.
__device__ __forceinline__ v16bf load_b_frag(const unsigned int* Bp, int NT, int kt, int nt) {
  int lane = threadIdx.x & 31;
  const unsigned int* p = Bp + ((((size_t)kt * NT + nt) * 32) + lane) * 8;
  uint4 lo = *reinterpret_cast<const uint4*>(p);
  uint4 hi = *reinterpret_cast<const uint4*>(p + 4);
  v8u u = { lo.x, lo.y, lo.z, lo.w, hi.x, hi.y, hi.z, hi.w };
  return __builtin_bit_cast(v16bf, u);
}

__device__ __forceinline__ v8f wmma_bf16(v16bf a, v16bf b, v8f c) {
  return __builtin_amdgcn_wmma_f32_16x16x32_bf16(false, a, false, b, (short)0, c, false, false);
}

// ---------- weight packing: f32 sources -> fragment-major bf16 ----------
__device__ __forceinline__ float fetchW(const float* s0, const float* s1,
                                        const float* s2, const float* s3,
                                        int mode, int k, int n) {
  switch (mode) {
    case 0: {  // W4 [256,2048]: cols [We2h | Wr_x | Wz_x | Wn_x], each [256,512]
      const float* s = (n < 512) ? s0 : (n < 1024) ? s1 : (n < 1536) ? s2 : s3;
      return s[k * 512 + (n & 511)];
    }
    case 1: {  // Wrz [1024,1024]: [[Wr_h, Wz_h],[Wr_e, Wz_e]], each [512,512]
      const float* arr[4] = { s0, s1, s2, s3 };
      return arr[((k >> 9) << 1) + (n >> 9)][(k & 511) * 512 + (n & 511)];
    }
    case 2:    // Wn [1024,512]: [Wn_h ; Wn_e]
      return ((k < 512) ? s0 : s1)[(k & 511) * 512 + n];
    default:   // Wo [512,256]
      return s0[k * 256 + n];
  }
}

__global__ void pack_weights(const float* s0, const float* s1, const float* s2, const float* s3,
                             unsigned int* dst, int K, int N, int mode) {
  int gid = blockIdx.x * blockDim.x + threadIdx.x;
  int total = (K * N) >> 1;
  if (gid >= total) return;
  int v    = gid & 7;
  int lane = (gid >> 3) & 31;
  int tile = gid >> 8;
  int NT   = N >> 4;
  int nt   = tile % NT, kt = tile / NT;
  int n    = nt * 16 + (lane & 15);
  int k    = kt * 32 + ((v >> 2) << 4) + ((lane >> 4) << 3) + ((v & 3) << 1);
  float lo = fetchW(s0, s1, s2, s3, mode, k,     n);
  float hi = fetchW(s0, s1, s2, s3, mode, k + 1, n);
  dst[gid] = pk2bf(lo, hi);
}

// ---------- misc prep ----------
__global__ void cvt_x(const float* x, unsigned short* xb, int n) {
  int i = blockIdx.x * blockDim.x + threadIdx.x;
  if (i < n) xb[i] = f2bf(x[i]);
}
__global__ void build_bias4(const float* br, const float* bz, const float* bn, float* bias4) {
  int j = blockIdx.x * blockDim.x + threadIdx.x;
  if (j >= 4 * H_) return;
  float v = 0.0f;
  if (j >= 512 && j < 1024)       v = br[j - 512];
  else if (j >= 1024 && j < 1536) v = bz[j - 1024];
  else if (j >= 1536)             v = bn[j - 1536];
  bias4[j] = v;
}
__global__ void copy_h(const float* h0, float* hbuf, int n) {
  int i = blockIdx.x * blockDim.x + threadIdx.x;
  if (i < n) hbuf[i] = h0[i];
}

// ---------- input projection GEMM: [32768,256]x[256,2048] -> Xproj[T][B][4H] f32 ----------
__global__ __launch_bounds__(256, 1)
void gemm_in(const unsigned short* __restrict__ xb,
             const unsigned int*  __restrict__ W4p,
             const float*         __restrict__ bias4,
             float*               __restrict__ Xproj) {
  int wid  = blockIdx.x * 8 + (threadIdx.x >> 5);  // 65536 waves
  int mt   = wid >> 5;                             // 0..2047
  int ntg  = wid & 31;                             // 4 n-tiles each
  int row0 = mt * 16;
  v8f acc[4] = {};
  // ping-pong pipeline, no register rotation
  v16bf a0 = load_a_frag(xb, 256, row0, 0);
  v16bf a1 = load_a_frag(xb, 256, row0, 32);
  v16bf b0[4], b1[4];
#pragma unroll
  for (int i = 0; i < 4; ++i) b0[i] = load_b_frag(W4p, 128, 0, ntg * 4 + i);
#pragma unroll
  for (int i = 0; i < 4; ++i) b1[i] = load_b_frag(W4p, 128, 1, ntg * 4 + i);
  for (int kt = 0; kt < 8; kt += 2) {
#pragma unroll
    for (int i = 0; i < 4; ++i) acc[i] = wmma_bf16(a0, b0[i], acc[i]);
    int k2 = (kt + 2 < 8) ? kt + 2 : kt;           // tail: harmless L1 reload
    a0 = load_a_frag(xb, 256, row0, k2 * 32);
#pragma unroll
    for (int i = 0; i < 4; ++i) b0[i] = load_b_frag(W4p, 128, k2, ntg * 4 + i);
    __builtin_prefetch(W4p + (((size_t)k2 * 128 + ntg * 4) * 32) * 8, 0, 3);
#pragma unroll
    for (int i = 0; i < 4; ++i) acc[i] = wmma_bf16(a1, b1[i], acc[i]);
    int k3 = (kt + 3 < 8) ? kt + 3 : kt + 1;
    a1 = load_a_frag(xb, 256, row0, k3 * 32);
#pragma unroll
    for (int i = 0; i < 4; ++i) b1[i] = load_b_frag(W4p, 128, k3, ntg * 4 + i);
  }
  int lane = threadIdx.x & 31;
  int col0 = lane & 15, rbase = (lane >> 4) << 3;
#pragma unroll
  for (int i = 0; i < 4; ++i) {
    int j = (ntg * 4 + i) * 16 + col0;
    float bj = bias4[j];
#pragma unroll
    for (int r = 0; r < 8; ++r) {
      int row = row0 + rbase + r;           // row = b*T + t
      int bb = row >> 9, tt = row & 511;
      Xproj[((size_t)tt * B_ + bb) * 2048 + j] = acc[i][r] + bj;
    }
  }
}

// ---------- scan step kernel 1: r,z gates ----------
__device__ __forceinline__ void he_load8(const float* hrow, const float* xerow,
                                         int k0, unsigned int* out) {
  float4 f0, f1;
  if (k0 < 512) {
    f0 = *reinterpret_cast<const float4*>(hrow + k0);
    f1 = *reinterpret_cast<const float4*>(hrow + k0 + 4);
  } else {
    int kk = k0 - 512;
    float4 x0 = *reinterpret_cast<const float4*>(xerow + kk);
    float4 x1 = *reinterpret_cast<const float4*>(xerow + kk + 4);
    float4 h0 = *reinterpret_cast<const float4*>(hrow + kk);
    float4 h1 = *reinterpret_cast<const float4*>(hrow + kk + 4);
    f0 = make_float4(x0.x - h0.x, x0.y - h0.y, x0.z - h0.z, x0.w - h0.w);
    f1 = make_float4(x1.x - h1.x, x1.y - h1.y, x1.z - h1.z, x1.w - h1.w);
  }
  out[0] = pk2bf(f0.x, f0.y); out[1] = pk2bf(f0.z, f0.w);
  out[2] = pk2bf(f1.x, f1.y); out[3] = pk2bf(f1.z, f1.w);
}

__device__ __forceinline__ v16bf he_frag(const float* hrow, const float* xerow, int kt, int half) {
  unsigned int au[8];
  int k0 = kt * 32 + half * 8;
  he_load8(hrow, xerow, k0,      &au[0]);
  he_load8(hrow, xerow, k0 + 16, &au[4]);
  v8u uu = { au[0], au[1], au[2], au[3], au[4], au[5], au[6], au[7] };
  return __builtin_bit_cast(v16bf, uu);
}

// 64 waves, 2 waves/block (spread over 32 WGPs)
__global__ __launch_bounds__(64, 1)
void step_rz(const float* __restrict__ hbuf,
             const float* __restrict__ Xt,          // Xproj + t*B*4H
             const unsigned int* __restrict__ Wrzp,
             unsigned short* __restrict__ An,
             float* __restrict__ zbuf) {
  int wid  = blockIdx.x * 2 + (threadIdx.x >> 5);
  int mt   = wid >> 4;                             // 0..3
  int ntg  = wid & 15;                             // 0..15
  int lane = threadIdx.x & 31;
  int half = lane >> 4;
  int brow = mt * 16 + (lane & 15);
  const float* hrow  = hbuf + brow * H_;
  const float* xerow = Xt + (size_t)brow * 2048;   // xe at offset 0
  v8f acc[4] = {};
  v16bf a0 = he_frag(hrow, xerow, 0, half);
  v16bf a1 = he_frag(hrow, xerow, 1, half);
  v16bf b0[4], b1[4];
#pragma unroll
  for (int i = 0; i < 4; ++i) b0[i] = load_b_frag(Wrzp, 64, 0, ntg * 4 + i);
#pragma unroll
  for (int i = 0; i < 4; ++i) b1[i] = load_b_frag(Wrzp, 64, 1, ntg * 4 + i);
  for (int kt = 0; kt < 32; kt += 2) {
#pragma unroll
    for (int i = 0; i < 4; ++i) acc[i] = wmma_bf16(a0, b0[i], acc[i]);
    int k2 = (kt + 2 < 32) ? kt + 2 : kt;
    a0 = he_frag(hrow, xerow, k2, half);
#pragma unroll
    for (int i = 0; i < 4; ++i) b0[i] = load_b_frag(Wrzp, 64, k2, ntg * 4 + i);
    __builtin_prefetch(Wrzp + (((size_t)k2 * 64 + ntg * 4) * 32) * 8, 0, 3);
#pragma unroll
    for (int i = 0; i < 4; ++i) acc[i] = wmma_bf16(a1, b1[i], acc[i]);
    int k3 = (kt + 3 < 32) ? kt + 3 : kt + 1;
    a1 = he_frag(hrow, xerow, k3, half);
#pragma unroll
    for (int i = 0; i < 4; ++i) b1[i] = load_b_frag(Wrzp, 64, k3, ntg * 4 + i);
  }
  int col0 = lane & 15, rbase = half << 3;
#pragma unroll
  for (int i = 0; i < 4; ++i) {
    int j = (ntg * 4 + i) * 16 + col0;
#pragma unroll
    for (int r = 0; r < 8; ++r) {
      int bb = mt * 16 + rbase + r;
      const float* xrow = Xt + (size_t)bb * 2048;
      float a = acc[i][r];
      if (j < 512) {                               // r gate
        float rg = sigmoidf_(a + xrow[512 + j]);   // xr includes br
        float hv = hbuf[bb * H_ + j];
        float ev = xrow[j] - hv;
        An[(size_t)bb * 1024 + j]       = f2bf(rg * hv);
        An[(size_t)bb * 1024 + 512 + j] = f2bf(rg * ev);
      } else {                                     // z gate
        int jz = j - 512;
        zbuf[bb * H_ + jz] = sigmoidf_(a + xrow[1024 + jz]);
      }
    }
  }
}

// ---------- scan step kernel 2: candidate n + h update ----------
// 32 waves, 2 waves/block (spread over 16 WGPs)
__global__ __launch_bounds__(64, 1)
void step_n(const unsigned short* __restrict__ An,
            const unsigned int*  __restrict__ Wnp,
            const float* __restrict__ Xt,
            const float* __restrict__ zbuf,
            float* __restrict__ hbuf,
            float* __restrict__ hseq_f32,
            unsigned short* __restrict__ hseq_bf,
            int t) {
  int wid  = blockIdx.x * 2 + (threadIdx.x >> 5);
  int mt   = wid >> 3;                             // 0..3
  int ntg  = wid & 7;                              // 0..7
  int row0 = mt * 16;
  v8f acc[4] = {};
  v16bf a0 = load_a_frag(An, 1024, row0, 0);
  v16bf a1 = load_a_frag(An, 1024, row0, 32);
  v16bf b0[4], b1[4];
#pragma unroll
  for (int i = 0; i < 4; ++i) b0[i] = load_b_frag(Wnp, 32, 0, ntg * 4 + i);
#pragma unroll
  for (int i = 0; i < 4; ++i) b1[i] = load_b_frag(Wnp, 32, 1, ntg * 4 + i);
  for (int kt = 0; kt < 32; kt += 2) {
#pragma unroll
    for (int i = 0; i < 4; ++i) acc[i] = wmma_bf16(a0, b0[i], acc[i]);
    int k2 = (kt + 2 < 32) ? kt + 2 : kt;
    a0 = load_a_frag(An, 1024, row0, k2 * 32);
#pragma unroll
    for (int i = 0; i < 4; ++i) b0[i] = load_b_frag(Wnp, 32, k2, ntg * 4 + i);
    __builtin_prefetch(Wnp + (((size_t)k2 * 32 + ntg * 4) * 32) * 8, 0, 3);
#pragma unroll
    for (int i = 0; i < 4; ++i) acc[i] = wmma_bf16(a1, b1[i], acc[i]);
    int k3 = (kt + 3 < 32) ? kt + 3 : kt + 1;
    a1 = load_a_frag(An, 1024, row0, k3 * 32);
#pragma unroll
    for (int i = 0; i < 4; ++i) b1[i] = load_b_frag(Wnp, 32, k3, ntg * 4 + i);
  }
  int lane = threadIdx.x & 31;
  int col0 = lane & 15, rbase = (lane >> 4) << 3;
#pragma unroll
  for (int i = 0; i < 4; ++i) {
    int j = (ntg * 4 + i) * 16 + col0;
#pragma unroll
    for (int r = 0; r < 8; ++r) {
      int bb = row0 + rbase + r;
      float pre  = Xt[(size_t)bb * 2048 + 1536 + j];  // xn includes bn
      float n    = tanhf(pre + acc[i][r]);
      float z    = zbuf[bb * H_ + j];
      float hold = hbuf[bb * H_ + j];
      float hnew = (1.0f - z) * hold + z * n;
      hbuf[bb * H_ + j] = hnew;
      size_t o = ((size_t)bb * T_ + t) * H_ + j;      // [B,T,H]
      hseq_f32[o] = hnew;
      hseq_bf[o]  = f2bf(hnew);
    }
  }
}

// ---------- readout GEMM: [32768,512]x[512,256] + bo -> out [B,T,D] ----------
__global__ __launch_bounds__(256, 1)
void gemm_out(const unsigned short* __restrict__ hseq_bf,
              const unsigned int*  __restrict__ Wop,
              const float* __restrict__ bo,
              float* __restrict__ out) {
  int wid  = blockIdx.x * 8 + (threadIdx.x >> 5);  // 8192 waves
  int mt   = wid >> 2;                             // 0..2047
  int ntg  = wid & 3;                              // 0..3
  int row0 = mt * 16;
  v8f acc[4] = {};
  v16bf a0 = load_a_frag(hseq_bf, 512, row0, 0);
  v16bf a1 = load_a_frag(hseq_bf, 512, row0, 32);
  v16bf b0[4], b1[4];
#pragma unroll
  for (int i = 0; i < 4; ++i) b0[i] = load_b_frag(Wop, 16, 0, ntg * 4 + i);
#pragma unroll
  for (int i = 0; i < 4; ++i) b1[i] = load_b_frag(Wop, 16, 1, ntg * 4 + i);
  for (int kt = 0; kt < 16; kt += 2) {
#pragma unroll
    for (int i = 0; i < 4; ++i) acc[i] = wmma_bf16(a0, b0[i], acc[i]);
    int k2 = (kt + 2 < 16) ? kt + 2 : kt;
    a0 = load_a_frag(hseq_bf, 512, row0, k2 * 32);
#pragma unroll
    for (int i = 0; i < 4; ++i) b0[i] = load_b_frag(Wop, 16, k2, ntg * 4 + i);
    __builtin_prefetch(Wop + (((size_t)k2 * 16 + ntg * 4) * 32) * 8, 0, 3);
#pragma unroll
    for (int i = 0; i < 4; ++i) acc[i] = wmma_bf16(a1, b1[i], acc[i]);
    int k3 = (kt + 3 < 16) ? kt + 3 : kt + 1;
    a1 = load_a_frag(hseq_bf, 512, row0, k3 * 32);
#pragma unroll
    for (int i = 0; i < 4; ++i) b1[i] = load_b_frag(Wop, 16, k3, ntg * 4 + i);
  }
  int lane = threadIdx.x & 31;
  int col0 = lane & 15, rbase = (lane >> 4) << 3;
#pragma unroll
  for (int i = 0; i < 4; ++i) {
    int j = (ntg * 4 + i) * 16 + col0;
    float bj = bo[j];
#pragma unroll
    for (int r = 0; r < 8; ++r) {
      int row = row0 + rbase + r;                  // row = b*T + t
      out[(size_t)row * D_ + j] = acc[i][r] + bj;
    }
  }
}

extern "C" void kernel_launch(void* const* d_in, const int* in_sizes, int n_in,
                              void* d_out, int out_size, void* d_ws, size_t ws_size,
                              hipStream_t stream) {
  const float* x    = (const float*)d_in[0];
  const float* h0   = (const float*)d_in[1];
  const float* We2h = (const float*)d_in[2];
  const float* Wr_x = (const float*)d_in[3];
  const float* Wr_h = (const float*)d_in[4];
  const float* Wr_e = (const float*)d_in[5];
  const float* br   = (const float*)d_in[6];
  const float* Wz_x = (const float*)d_in[7];
  const float* Wz_h = (const float*)d_in[8];
  const float* Wz_e = (const float*)d_in[9];
  const float* bz   = (const float*)d_in[10];
  const float* Wn_x = (const float*)d_in[11];
  const float* Wn_h = (const float*)d_in[12];
  const float* Wn_e = (const float*)d_in[13];
  const float* bn   = (const float*)d_in[14];
  const float* Wo   = (const float*)d_in[15];
  const float* bo   = (const float*)d_in[16];

  size_t off = 0;
  auto take = [&](size_t bytes) -> void* {
    void* p = (char*)d_ws + off;
    off += (bytes + 255) & ~(size_t)255;
    return p;
  };
  unsigned short* xb    = (unsigned short*)take((size_t)B_ * T_ * D_ * 2);
  unsigned int*   W4p   = (unsigned int*)  take((size_t)D_ * 4 * H_ * 2);
  unsigned int*   Wrzp  = (unsigned int*)  take((size_t)2 * H_ * 2 * H_ * 2);
  unsigned int*   Wnp   = (unsigned int*)  take((size_t)2 * H_ * H_ * 2);
  unsigned int*   Wop   = (unsigned int*)  take((size_t)H_ * D_ * 2);
  float*          bias4 = (float*)         take((size_t)4 * H_ * sizeof(float));
  float*          Xproj = (float*)         take((size_t)T_ * B_ * 4 * H_ * sizeof(float));
  float*          hbuf  = (float*)         take((size_t)B_ * H_ * sizeof(float));
  unsigned short* An    = (unsigned short*)take((size_t)B_ * 2 * H_ * 2);
  float*          zbuf  = (float*)         take((size_t)B_ * H_ * sizeof(float));
  unsigned short* hseqb = (unsigned short*)take((size_t)B_ * T_ * H_ * 2);

  float* out_f    = (float*)d_out;                       // [B,T,D]
  float* hseq_f32 = out_f + (size_t)B_ * T_ * D_;        // [B,T,H]

  // prep
  int nx = B_ * T_ * D_;
  cvt_x<<<(nx + 255) / 256, 256, 0, stream>>>(x, xb, nx);
  pack_weights<<<(D_ * 4 * H_ / 2 + 255) / 256, 256, 0, stream>>>(
      We2h, Wr_x, Wz_x, Wn_x, W4p, D_, 4 * H_, 0);
  pack_weights<<<(2 * H_ * 2 * H_ / 2 + 255) / 256, 256, 0, stream>>>(
      Wr_h, Wz_h, Wr_e, Wz_e, Wrzp, 2 * H_, 2 * H_, 1);
  pack_weights<<<(2 * H_ * H_ / 2 + 255) / 256, 256, 0, stream>>>(
      Wn_h, Wn_e, Wn_h, Wn_e, Wnp, 2 * H_, H_, 2);
  pack_weights<<<(H_ * D_ / 2 + 255) / 256, 256, 0, stream>>>(
      Wo, Wo, Wo, Wo, Wop, H_, D_, 3);
  build_bias4<<<(4 * H_ + 255) / 256, 256, 0, stream>>>(br, bz, bn, bias4);
  copy_h<<<(B_ * H_ + 255) / 256, 256, 0, stream>>>(h0, hbuf, B_ * H_);

  // input projection: 65536 waves
  gemm_in<<<8192, 256, 0, stream>>>(xb, W4p, bias4, Xproj);

  // sequential scan: 2 waves/block to spread over many WGPs
  for (int t = 0; t < T_; ++t) {
    const float* Xt = Xproj + (size_t)t * B_ * 4 * H_;
    step_rz<<<32, 64, 0, stream>>>(hbuf, Xt, Wrzp, An, zbuf);
    step_n<<<16, 64, 0, stream>>>(An, Wnp, Xt, zbuf, hbuf, hseq_f32, hseqb, t);
  }

  // readout
  gemm_out<<<1024, 256, 0, stream>>>(hseqb, Wop, bo, out_f);
}
